// TransformerMemoryUpdater_5574867551057
// MI455X (gfx1250) — matrix operand order repaired and checked by reading, not compiled, not run on hardware
//
#include <hip/hip_runtime.h>
#include <hip/hip_bf16.h>
#include <cstdint>

#define NNODES 50000
#define MBOX   10
#define DIN    128
#define DOUT   128
#define DTIME  128
#define DKCAT  256   // DIN + DTIME

typedef __attribute__((ext_vector_type(16))) __bf16 v16bf;
typedef __attribute__((ext_vector_type(8)))  float  v8f;

// ---------------- LDS layout (bytes) ----------------
// One B-fragment = 32 lanes * 16 halves * 2B = 1024 B
#define OFF_WQ   0                      // 4kc*8ct = 32 frags -> 32 KB
#define OFF_WK   (OFF_WQ + 32 * 1024)   // 8kc*8ct = 64 frags -> 64 KB
#define OFF_WV   (OFF_WK + 64 * 1024)   // 64 KB
#define OFF_WM   (OFF_WV + 64 * 1024)   // 32 KB
#define OFF_NRM  (OFF_WM + 32 * 1024)   // 8 waves * 16*128 bf16 = 32 KB
#define OFF_VEC  (OFF_NRM + 32 * 1024)  // 8 * 128 f32 = 4 KB
#define OFF_Q    (OFF_VEC + 4096)       // 8 waves * 16*128 f32 = 64 KB
#define SMEM_BYTES (OFF_Q + 64 * 1024)  // 299,008 B  (< 320 KB WGP LDS)

// Stage a torch-layout weight W[out][kd] into LDS as bf16, pre-swizzled into
// WMMA B-fragment order for v_wmma_f32_16x16x32_bf16:
//   B element i of lane L in frag (kc,ct):  k = kc*32 + (L>=16 ? 8:0) + (i&7) + (i>=8 ? 16:0),
//                                           n = ct*16 + (L&15)
__device__ __forceinline__ void stage_frags(__bf16* dst, const float* __restrict__ W,
                                            int kd, int tid) {
  const int total = kd * 128;
  for (int idx = tid; idx < total; idx += 256) {
    const int i  = idx & 15;
    const int L  = (idx >> 4) & 31;
    const int ct = (idx >> 9) & 7;
    const int kc = idx >> 12;
    const int k  = kc * 32 + ((L & 16) ? 8 : 0) + (i & 7) + ((i & 8) ? 16 : 0);
    const int o  = ct * 16 + (L & 15);
    dst[idx] = (__bf16)W[o * kd + k];
  }
}

__device__ __forceinline__ v16bf ldB(const __bf16* fr, int kc, int ct, int lane) {
  return *(const v16bf*)(fr + ((((kc << 3) + ct) << 9) + (lane << 4)));
}

// Build a 16-half A-fragment slice from 32 contiguous f32 cols (chunk start p):
// halves 0..7 = cols [0,8), halves 8..15 = cols [16,24) (caller pre-adds half*8).
__device__ __forceinline__ v16bf packA16(const float* p) {
  const float4 f0 = *(const float4*)(p);
  const float4 f1 = *(const float4*)(p + 4);
  const float4 f2 = *(const float4*)(p + 16);
  const float4 f3 = *(const float4*)(p + 20);
  v16bf a;
  a[0]  = (__bf16)f0.x; a[1]  = (__bf16)f0.y; a[2]  = (__bf16)f0.z; a[3]  = (__bf16)f0.w;
  a[4]  = (__bf16)f1.x; a[5]  = (__bf16)f1.y; a[6]  = (__bf16)f1.z; a[7]  = (__bf16)f1.w;
  a[8]  = (__bf16)f2.x; a[9]  = (__bf16)f2.y; a[10] = (__bf16)f2.z; a[11] = (__bf16)f2.w;
  a[12] = (__bf16)f3.x; a[13] = (__bf16)f3.y; a[14] = (__bf16)f3.z; a[15] = (__bf16)f3.w;
  return a;
}

#define WMMA_BF16(A, B, C) \
  __builtin_amdgcn_wmma_f32_16x16x32_bf16(false, (A), false, (B), (short)0, (C), false, false)

__global__ void __launch_bounds__(256)
tmu_kernel(const float* __restrict__ mem,      const float* __restrict__ mem_input,
           const float* __restrict__ ts,       const float* __restrict__ mail_ts,
           const float* __restrict__ wq_w,     const float* __restrict__ wq_b,
           const float* __restrict__ wk_w,     const float* __restrict__ wk_b,
           const float* __restrict__ wv_w,     const float* __restrict__ wv_b,
           const float* __restrict__ ln_g,     const float* __restrict__ ln_b,
           const float* __restrict__ mlp_w,    const float* __restrict__ mlp_b,
           const float* __restrict__ te_w,     const float* __restrict__ te_b,
           float* __restrict__ out) {
  extern __shared__ __align__(16) char smem[];
  __bf16* fWQ    = (__bf16*)(smem + OFF_WQ);
  __bf16* fWK    = (__bf16*)(smem + OFF_WK);
  __bf16* fWV    = (__bf16*)(smem + OFF_WV);
  __bf16* fWM    = (__bf16*)(smem + OFF_WM);
  __bf16* nrmAll = (__bf16*)(smem + OFF_NRM);
  float*  vec    = (float*)(smem + OFF_VEC);
  float*  qAll   = (float*)(smem + OFF_Q);

  const int tid = threadIdx.x;
  stage_frags(fWQ, wq_w, 128, tid);
  stage_frags(fWK, wk_w, 256, tid);
  stage_frags(fWV, wv_w, 256, tid);
  stage_frags(fWM, mlp_w, 128, tid);
  if (tid < 128) {
    vec[0 * 128 + tid] = te_w[tid];
    vec[1 * 128 + tid] = te_b[tid];
    vec[2 * 128 + tid] = wq_b[tid];
    vec[3 * 128 + tid] = wk_b[tid];
    vec[4 * 128 + tid] = wv_b[tid];
    vec[5 * 128 + tid] = ln_g[tid];
    vec[6 * 128 + tid] = ln_b[tid];
    vec[7 * 128 + tid] = mlp_b[tid];
  }
  __syncthreads();

  const int lane = tid & 31;
  const int wid  = tid >> 5;
  const int half = lane >> 4;   // which 16-lane half of the wave
  const int l15  = lane & 15;
  float*  ldsQ   = qAll   + wid * (16 * 128);
  __bf16* ldsNrm = nrmAll + wid * (16 * 128);
  const float* teW = vec;
  const float* teB = vec + 128;
  const float* qB  = vec + 256;
  const float* kB  = vec + 384;
  const float* vB  = vec + 512;
  const float* lnG = vec + 640;
  const float* lnB = vec + 768;
  const float* mB  = vec + 896;

  const int NB = (NNODES + 15) >> 4;
  for (int b = blockIdx.x * 8 + wid; b < NB; b += gridDim.x * 8) {
    const int base = b << 4;

    // ---------------- Q-tile [16 nodes x 128] = mem @ Wq^T + bq ----------------
    {
      v8f accQ[8];
#pragma unroll
      for (int ct = 0; ct < 8; ++ct)
#pragma unroll
        for (int r = 0; r < 8; ++r) accQ[ct][r] = 0.f;
      int qrow = base + l15; if (qrow >= NNODES) qrow = NNODES - 1;
      const float* mrow = mem + (size_t)qrow * DOUT;
#pragma unroll
      for (int kc = 0; kc < 4; ++kc) {
        const v16bf a = packA16(mrow + kc * 32 + half * 8);
#pragma unroll
        for (int ct = 0; ct < 8; ++ct)
          accQ[ct] = WMMA_BF16(a, ldB(fWQ, kc, ct, lane), accQ[ct]);
      }
#pragma unroll
      for (int ct = 0; ct < 8; ++ct) {
        const float bq = qB[ct * 16 + l15];
#pragma unroll
        for (int r = 0; r < 8; ++r)
          ldsQ[(r + half * 8) * 128 + ct * 16 + l15] = accQ[ct][r] + bq;
      }
    }

    // ---------------- per node: K/V projections (WMMA) + attention ----------------
    for (int ln = 0; ln < 16; ++ln) {
      const int n = base + ln;
      if (n >= NNODES) break;
      if (n + 1 < NNODES)  // global_prefetch_b8: next node's mailbox (lanes fan out)
        __builtin_prefetch(mem_input + (size_t)(n + 1) * (MBOX * DIN) + lane * 40, 0, 1);

      const int  m     = l15;            // A-tile row = mail index (rows 10..15 pad)
      const bool vld   = (m < MBOX);
      const float tsn  = ts[n];
      const float dtm  = vld ? (tsn - mail_ts[(size_t)n * MBOX + m]) : 0.f;
      const float* mirow = mem_input + (size_t)n * (MBOX * DIN) + m * DIN;

      v8f accK[8], accV[8];
#pragma unroll
      for (int ct = 0; ct < 8; ++ct)
#pragma unroll
        for (int r = 0; r < 8; ++r) { accK[ct][r] = 0.f; accV[ct][r] = 0.f; }

#pragma unroll
      for (int kc = 0; kc < 8; ++kc) {
        v16bf a;
        if (kc < 4) {                       // message part of concat
          if (vld) {
            a = packA16(mirow + kc * 32 + half * 8);
          } else {
#pragma unroll
            for (int i = 0; i < 16; ++i) a[i] = (__bf16)0.f;
          }
        } else {                            // time-encoding part: cos(dt*w + b)
          const int j0 = (kc - 4) * 32 + half * 8;
          const float4 w0 = *(const float4*)(teW + j0);
          const float4 w1 = *(const float4*)(teW + j0 + 4);
          const float4 w2 = *(const float4*)(teW + j0 + 16);
          const float4 w3 = *(const float4*)(teW + j0 + 20);
          const float4 b0 = *(const float4*)(teB + j0);
          const float4 b1 = *(const float4*)(teB + j0 + 4);
          const float4 b2 = *(const float4*)(teB + j0 + 16);
          const float4 b3 = *(const float4*)(teB + j0 + 20);
          a[0]  = (__bf16)__cosf(dtm * w0.x + b0.x);
          a[1]  = (__bf16)__cosf(dtm * w0.y + b0.y);
          a[2]  = (__bf16)__cosf(dtm * w0.z + b0.z);
          a[3]  = (__bf16)__cosf(dtm * w0.w + b0.w);
          a[4]  = (__bf16)__cosf(dtm * w1.x + b1.x);
          a[5]  = (__bf16)__cosf(dtm * w1.y + b1.y);
          a[6]  = (__bf16)__cosf(dtm * w1.z + b1.z);
          a[7]  = (__bf16)__cosf(dtm * w1.w + b1.w);
          a[8]  = (__bf16)__cosf(dtm * w2.x + b2.x);
          a[9]  = (__bf16)__cosf(dtm * w2.y + b2.y);
          a[10] = (__bf16)__cosf(dtm * w2.z + b2.z);
          a[11] = (__bf16)__cosf(dtm * w2.w + b2.w);
          a[12] = (__bf16)__cosf(dtm * w3.x + b3.x);
          a[13] = (__bf16)__cosf(dtm * w3.y + b3.y);
          a[14] = (__bf16)__cosf(dtm * w3.z + b3.z);
          a[15] = (__bf16)__cosf(dtm * w3.w + b3.w);
          if (!vld) {
#pragma unroll
            for (int i = 0; i < 16; ++i) a[i] = (__bf16)0.f;
          }
        }
#pragma unroll
        for (int ct = 0; ct < 8; ++ct) {
          accK[ct] = WMMA_BF16(a, ldB(fWK, kc, ct, lane), accK[ct]);
          accV[ct] = WMMA_BF16(a, ldB(fWV, kc, ct, lane), accV[ct]);
        }
      }

      // ---- attention per head (col-tile ct == head), softmax over mailbox ----
      float rstv[8];
#pragma unroll
      for (int ct = 0; ct < 8; ++ct) {
        const float kb = kB[ct * 16 + l15];
        const float vb = vB[ct * 16 + l15];
        const float qv = ldsQ[ln * 128 + ct * 16 + l15];
        float att[8], oth[8], e[8];
#pragma unroll
        for (int r = 0; r < 8; ++r) {
          float p = (accK[ct][r] + kb) * qv;           // Q[h,d]*K[m,h,d] per lane
          p += __shfl_xor(p, 1);
          p += __shfl_xor(p, 2);
          p += __shfl_xor(p, 4);
          p += __shfl_xor(p, 8);                       // 16-lane dot within half
          p = (p > 0.f) ? p : 0.2f * p;                // LeakyReLU
          att[r] = ((r + half * 8) < MBOX) ? p : -3.0e38f;
        }
#pragma unroll
        for (int r = 0; r < 8; ++r) oth[r] = __shfl_xor(att[r], 16);
        float mx = -3.0e38f;
#pragma unroll
        for (int r = 0; r < 8; ++r) { mx = fmaxf(mx, att[r]); mx = fmaxf(mx, oth[r]); }
        float ssum = 0.f;
#pragma unroll
        for (int r = 0; r < 8; ++r) {
          e[r]  = __expf(att[r] - mx);
          ssum += e[r] + __expf(oth[r] - mx);
        }
        const float inv = 1.f / ssum;
        float acc = 0.f;
#pragma unroll
        for (int r = 0; r < 8; ++r) acc += (e[r] * inv) * (accV[ct][r] + vb);
        acc += __shfl_xor(acc, 16);                    // merge the two mail halves
        acc += mem[(size_t)n * DOUT + ct * 16 + l15];  // residual
        rstv[ct] = acc;
      }

      // ---- LayerNorm across 128 cols, fully in registers ----
      float su = 0.f;
#pragma unroll
      for (int ct = 0; ct < 8; ++ct) su += rstv[ct];
      su += __shfl_xor(su, 1); su += __shfl_xor(su, 2);
      su += __shfl_xor(su, 4); su += __shfl_xor(su, 8);
      const float mu = su * (1.f / 128.f);
      float s2 = 0.f;
#pragma unroll
      for (int ct = 0; ct < 8; ++ct) { const float d = rstv[ct] - mu; s2 += d * d; }
      s2 += __shfl_xor(s2, 1); s2 += __shfl_xor(s2, 2);
      s2 += __shfl_xor(s2, 4); s2 += __shfl_xor(s2, 8);
      const float rs = rsqrtf(s2 * (1.f / 128.f) + 1e-5f);
#pragma unroll
      for (int ct = 0; ct < 8; ++ct) {
        const int c = ct * 16 + l15;
        const float nv = (rstv[ct] - mu) * rs * lnG[c] + lnB[c];
        if (half == 0) ldsNrm[ln * 128 + c] = (__bf16)nv;   // stage for MLP WMMA
      }
    }  // node loop

    // ---------------- MLP for the 16-node tile (WMMA) + ReLU + store ----------------
    {
      v8f accM[8];
#pragma unroll
      for (int ct = 0; ct < 8; ++ct)
#pragma unroll
        for (int r = 0; r < 8; ++r) accM[ct][r] = 0.f;
      const __bf16* nr = ldsNrm + l15 * 128;
#pragma unroll
      for (int kc = 0; kc < 4; ++kc) {
        v16bf a;
        const __bf16* p = nr + kc * 32 + half * 8;
#pragma unroll
        for (int i = 0; i < 8; ++i) { a[i] = p[i]; a[i + 8] = p[16 + i]; }
#pragma unroll
        for (int ct = 0; ct < 8; ++ct)
          accM[ct] = WMMA_BF16(a, ldB(fWM, kc, ct, lane), accM[ct]);
      }
#pragma unroll
      for (int ct = 0; ct < 8; ++ct) {
        const float bb = mB[ct * 16 + l15];
#pragma unroll
        for (int r = 0; r < 8; ++r) {
          const int orow = base + r + half * 8;
          if (orow < NNODES) {
            float v = accM[ct][r] + bb;
            out[(size_t)orow * DOUT + ct * 16 + l15] = (v > 0.f) ? v : 0.f;
          }
        }
      }
    }
  }  // batch loop
}

extern "C" void kernel_launch(void* const* d_in, const int* in_sizes, int n_in,
                              void* d_out, int out_size, void* d_ws, size_t ws_size,
                              hipStream_t stream) {
  (void)in_sizes; (void)n_in; (void)out_size; (void)d_ws; (void)ws_size;
  const float* mem       = (const float*)d_in[0];
  const float* mem_input = (const float*)d_in[1];
  const float* ts        = (const float*)d_in[2];
  const float* mail_ts   = (const float*)d_in[3];
  const float* wq_w      = (const float*)d_in[4];
  const float* wq_b      = (const float*)d_in[5];
  const float* wk_w      = (const float*)d_in[6];
  const float* wk_b      = (const float*)d_in[7];
  const float* wv_w      = (const float*)d_in[8];
  const float* wv_b      = (const float*)d_in[9];
  const float* ln_g      = (const float*)d_in[10];
  const float* ln_b      = (const float*)d_in[11];
  const float* mlp_w     = (const float*)d_in[12];
  const float* mlp_b     = (const float*)d_in[13];
  const float* te_w      = (const float*)d_in[14];
  const float* te_b      = (const float*)d_in[15];
  float* out = (float*)d_out;

  hipFuncSetAttribute(reinterpret_cast<const void*>(tmu_kernel),
                      hipFuncAttributeMaxDynamicSharedMemorySize, SMEM_BYTES);
  dim3 grid(160), block(256);
  hipLaunchKernelGGL(tmu_kernel, grid, block, SMEM_BYTES, stream,
                     mem, mem_input, ts, mail_ts,
                     wq_w, wq_b, wk_w, wk_b, wv_w, wv_b,
                     ln_g, ln_b, mlp_w, mlp_b, te_w, te_b, out);
}
